// GIKT_53515292508602
// MI455X (gfx1250) — matrix operand order repaired and checked by reading, not compiled, not run on hardware
//
#include <hip/hip_runtime.h>
#include <hip/hip_bf16.h>

// GIKT on MI455X (gfx1250, wave32). Heavy math via v_wmma_f32_16x16x32_f16.
// LDS layouts are fragment-native so all LDS traffic is ds_load_b128 /
// ds_store_b128. This revision batch-loads the 8 B-fragments of each k-step
// before issuing the 8 WMMAs, so the scheduler can overlap LDS latency with
// matrix math (staggered s_wait_dscnt instead of dscnt==0 per WMMA).

typedef __attribute__((ext_vector_type(16))) _Float16 v16h;
typedef __attribute__((ext_vector_type(8)))  _Float16 v8h;
typedef __attribute__((ext_vector_type(8)))  float    v8f;

#define LDA 136  // padded half-stride (16B-aligned rows, conflict-free banks)

// ---------------------------------------------------------------- WMMA frags
// A 16x32 f16 (ISA 7.12.2): lane<16 rows M=lane; VGPR v<4 -> K=2v+8lg,
// v>=4 -> 16+2(v-4)+8lg. Per lane: contiguous K runs [k0+8lg,+8) and
// [k0+16+8lg,+8)  => two aligned 16B LDS loads.
__device__ __forceinline__ v16h frag_a_lds(const _Float16* A, int k0) {
  const int lane = threadIdx.x & 31;
  const int lg = lane >> 4, mr = lane & 15;
  const _Float16* p = A + mr * LDA + k0 + 8 * lg;
  const v8h lo = *(const v8h*)p;
  const v8h hi = *(const v8h*)(p + 16);
  return __builtin_shufflevector(lo, hi, 0, 1, 2, 3, 4, 5, 6, 7, 8, 9, 10, 11,
                                 12, 13, 14, 15);
}
// B 32x16 f16: lane holds column n = n0 + (lane&15), K runs [k0+8lg,+8) and
// [k0+16+8lg,+8) (same as A). With W stored column-major (Wt[n*LDA + k]) a
// lane's fragment is two 16B runs => two aligned 16B LDS loads.
__device__ __forceinline__ v16h frag_b_ldsT(const _Float16* Wt, int k0, int n0) {
  const int lane = threadIdx.x & 31;
  const int lg = lane >> 4, nc = lane & 15;
  const _Float16* p = Wt + (n0 + nc) * LDA + k0 + 8 * lg;
  const v8h lo = *(const v8h*)p;
  const v8h hi = *(const v8h*)(p + 16);
  return __builtin_shufflevector(lo, hi, 0, 1, 2, 3, 4, 5, 6, 7, 8, 9, 10, 11,
                                 12, 13, 14, 15);
}

__device__ __forceinline__ v8f wmma_f16(v16h a, v16h b, v8f c) {
  return __builtin_amdgcn_wmma_f32_16x16x32_f16(false, a, false, b, (short)0, c,
                                                false, false);
}

__device__ __forceinline__ float sigmoidf_(float x) {
  return 1.0f / (1.0f + __expf(-x));
}

// ---------------------------------------------------------------- generic GEMM
// Y[M x 128] = act( A[M x 128] @ W[128 x 128] + bias + tab[sel[row]] )
// A-tile built in LDS per mode:
//  0: from X (row stride ldx)
//  1: hop-2 gather: emb_q[l2] + mean4(emb_s[q_nb[l2]])     (M = 6400*40)
//  2: hop-1 gather: emb_s[l1] + mean10(emb_q[s_nb[l1]])    (M = 6400*4)
//  3: hop-0 gather: emb_q[q]  + mean4(emb_s[q_nb[q]])      (M = 6400)
//  4: combine:      P0[r] + mean_grp(P1[r*grp + j])
__global__ void __launch_bounds__(128)
k_gemm(const float* __restrict__ X, int ldx,
       const float* __restrict__ W, int ldw,
       const float* __restrict__ bias,
       const float* __restrict__ tab, int ldtab, const int* __restrict__ sel,
       float* __restrict__ Y, int M, int act, int mode,
       const int* __restrict__ question, const int* __restrict__ q_nb,
       const int* __restrict__ s_nb,
       const float* __restrict__ emb_q, const float* __restrict__ emb_s,
       const float* __restrict__ P0, const float* __restrict__ P1, int grp) {
  __shared__ __attribute__((aligned(16))) _Float16 Wl[128 * LDA];   // col-major
  __shared__ __attribute__((aligned(16))) _Float16 Al[4][16 * LDA]; // row-major
  const int tid = threadIdx.x;

  // stage W column-major: chunk cid -> (n = cid&127, kc = cid>>7), 8 k's.
  // Global reads stay coalesced (consecutive lanes -> consecutive n).
  for (int it = 0; it < 16; ++it) {
    const int cid = tid + it * 128;
    const int n = cid & 127, kc = (cid >> 7) * 8;
    v8h hv;
#pragma unroll
    for (int j = 0; j < 8; ++j) hv[j] = (_Float16)W[(kc + j) * ldw + n];
    *(v8h*)&Wl[n * LDA + kc] = hv;
  }

  const int wv = tid >> 5, lane = tid & 31;
  const int tileM = (blockIdx.x * 4 + wv) * 16;
  // stage A: 16 rows x 16 chunks of 8 halves; indices computed per chunk.
  for (int it = 0; it < 8; ++it) {
    const int cid = lane + it * 32;           // 0..255
    const int r = cid >> 4, c8 = (cid & 15) * 8;
    const int gr = tileM + r;
    float f[8];
#pragma unroll
    for (int j = 0; j < 8; ++j) f[j] = 0.f;
    if (gr < M) {
      if (mode == 0) {
        const float* p = X + (long)gr * ldx + c8;
#pragma unroll
        for (int j = 0; j < 8; ++j) f[j] = p[j];
      } else if (mode == 1) {
        const int pos = gr / 40, t = gr % 40, n1 = t / 10, n2 = t % 10;
        const int q  = question[pos];
        const int l1 = q_nb[q * 4 + n1];
        const int l2 = s_nb[l1 * 10 + n2];
        const float* pq = emb_q + (long)l2 * 128 + c8;
        const float* p0 = emb_s + (long)q_nb[l2 * 4 + 0] * 128 + c8;
        const float* p1 = emb_s + (long)q_nb[l2 * 4 + 1] * 128 + c8;
        const float* p2 = emb_s + (long)q_nb[l2 * 4 + 2] * 128 + c8;
        const float* p3 = emb_s + (long)q_nb[l2 * 4 + 3] * 128 + c8;
#pragma unroll
        for (int j = 0; j < 8; ++j)
          f[j] = pq[j] + 0.25f * (p0[j] + p1[j] + p2[j] + p3[j]);
      } else if (mode == 2) {
        const int pos = gr / 4, n1 = gr % 4;
        const int q  = question[pos];
        const int l1 = q_nb[q * 4 + n1];
        const float* pb = emb_s + (long)l1 * 128 + c8;
        const float* pn[10];
#pragma unroll
        for (int u = 0; u < 10; ++u)
          pn[u] = emb_q + (long)s_nb[l1 * 10 + u] * 128 + c8;
#pragma unroll
        for (int j = 0; j < 8; ++j) {
          float s = 0.f;
#pragma unroll
          for (int u = 0; u < 10; ++u) s += pn[u][j];
          f[j] = pb[j] + 0.1f * s;
        }
      } else if (mode == 3) {
        const int q = question[gr];
        const float* pq = emb_q + (long)q * 128 + c8;
        const float* p0 = emb_s + (long)q_nb[q * 4 + 0] * 128 + c8;
        const float* p1 = emb_s + (long)q_nb[q * 4 + 1] * 128 + c8;
        const float* p2 = emb_s + (long)q_nb[q * 4 + 2] * 128 + c8;
        const float* p3 = emb_s + (long)q_nb[q * 4 + 3] * 128 + c8;
#pragma unroll
        for (int j = 0; j < 8; ++j)
          f[j] = pq[j] + 0.25f * (p0[j] + p1[j] + p2[j] + p3[j]);
      } else {  // mode 4
        const float* pb = P0 + (long)gr * 128 + c8;
        const float inv = 1.f / (float)grp;
#pragma unroll
        for (int j = 0; j < 8; ++j) f[j] = pb[j];
        for (int u = 0; u < grp; ++u) {
          const float* pp = P1 + ((long)gr * grp + u) * 128 + c8;
#pragma unroll
          for (int j = 0; j < 8; ++j) f[j] += inv * pp[j];
        }
      }
    }
    v8h hv;
#pragma unroll
    for (int j = 0; j < 8; ++j) hv[j] = (_Float16)f[j];
    *(v8h*)&Al[wv][r * LDA + c8] = hv;
  }
  __syncthreads();

  v8f zero = {0.f, 0.f, 0.f, 0.f, 0.f, 0.f, 0.f, 0.f};
  v8f acc[8];
#pragma unroll
  for (int nt = 0; nt < 8; ++nt) acc[nt] = zero;

#pragma unroll
  for (int ks = 0; ks < 4; ++ks) {
    const v16h a = frag_a_lds(&Al[wv][0], ks * 32);
    v16h bf[8];
#pragma unroll
    for (int nt = 0; nt < 8; ++nt) bf[nt] = frag_b_ldsT(Wl, ks * 32, nt * 16);
#pragma unroll
    for (int nt = 0; nt < 8; ++nt) acc[nt] = wmma_f16(a, bf[nt], acc[nt]);
  }

  const int lg = lane >> 4, nc = lane & 15;
#pragma unroll
  for (int nt = 0; nt < 8; ++nt) {
    const int n = nt * 16 + nc;
    const float bi = bias ? bias[n] : 0.f;
#pragma unroll
    for (int j = 0; j < 8; ++j) {
      const int m = tileM + 8 * lg + j;
      if (m < M) {
        float v = acc[nt][j] + bi;
        if (tab) v += tab[(long)sel[m] * ldtab + n];
        if (act == 1) v = tanhf(v);
        else if (act == 2) v = fmaxf(v, 0.f);
        Y[(long)m * 128 + n] = v;
      }
    }
  }
}

// ------------------------------------------------- r-side gate table (2x512)
__global__ void k_rcontrib(const float* __restrict__ emb_r,
                           const float* __restrict__ W_ih,
                           const float* __restrict__ b_ih,
                           const float* __restrict__ b_hh,
                           float* __restrict__ rcon) {
  const int idx = blockIdx.x * 256 + threadIdx.x;
  if (idx >= 1024) return;
  const int rv = idx >> 9, n = idx & 511;
  float s = b_ih[n] + b_hh[n];
  for (int k = 0; k < 128; ++k) s += emb_r[rv * 128 + k] * W_ih[(128 + k) * 512 + n];
  rcon[idx] = s;
}

// ----------------------------------------------------------- LSTM nonlinearity
__global__ void k_lstm(const float* __restrict__ Gi, const float* __restrict__ Gg,
                       const float* __restrict__ Go, float* __restrict__ h, int n) {
  const int i = blockIdx.x * 256 + threadIdx.x;
  if (i >= n) return;
  const float c = sigmoidf_(Gi[i]) * tanhf(Gg[i]);
  h[i] = sigmoidf_(Go[i]) * tanhf(c);
}

// ------------------------------------------------------- qs gather (B,199,5,D)
__global__ void k_qsflat(const int* __restrict__ question,
                         const int* __restrict__ qs_skills,
                         const float* __restrict__ emb_q,
                         const float* __restrict__ emb_s,
                         float* __restrict__ qsf) {
  const long i = (long)blockIdx.x * 256 + threadIdx.x;
  if (i >= (long)31840 * 128) return;
  const int c = (int)(i & 127);
  const long r = i >> 7;
  const int b = (int)(r / 995), rem = (int)(r % 995);
  const int t = rem / 5, m = rem % 5;
  const int qn = question[b * 200 + t + 1];
  const float* src = (m == 0) ? (emb_q + (long)qn * 128)
                              : (emb_s + (long)qs_skills[qn * 4 + (m - 1)] * 128);
  qsf[i] = src[c];
}

// -------------------------------------------- attention functionals v = W @ w
__global__ void k_wvec(const float* __restrict__ qW, const float* __restrict__ qb,
                       const float* __restrict__ kW, const float* __restrict__ kb,
                       const float* __restrict__ wW,
                       float* __restrict__ vq, float* __restrict__ vk,
                       float* __restrict__ cq, float* __restrict__ ck) {
  const int t = threadIdx.x;
  if (t < 128) {
    float s = 0.f;
    for (int d = 0; d < 128; ++d) s += qW[t * 128 + d] * wW[d];
    vq[t] = s;
  } else {
    const int u = t - 128;
    float s = 0.f;
    for (int d = 0; d < 128; ++d) s += kW[u * 128 + d] * wW[128 + d];
    vk[u] = s;
  }
  if (t == 0) { float s = 0.f; for (int d = 0; d < 128; ++d) s += qb[d] * wW[d]; *cq = s; }
  if (t == 1) { float s = 0.f; for (int d = 0; d < 128; ++d) s += kb[d] * wW[128 + d]; *ck = s; }
}

// out[r] = X[map(r)] . v + c  (one wave per row, shuffle reduce)
__global__ void __launch_bounds__(128)
k_rowdot(const float* __restrict__ X, const float* __restrict__ v,
         const float* __restrict__ cptr, float* __restrict__ out,
         int nrows, int statemap) {
  const int gw = blockIdx.x * 4 + (threadIdx.x >> 5);
  if (gw >= nrows) return;
  const int lane = threadIdx.x & 31;
  const long row = statemap ? ((long)(gw / 199) * 200 + (gw % 199)) : (long)gw;
  float s = 0.f;
  for (int d = lane; d < 128; d += 32) s += X[row * 128 + d] * v[d];
#pragma unroll
  for (int o = 16; o; o >>= 1) s += __shfl_xor(s, o, 32);
  if (lane == 0) out[gw] = s + *cptr;
}

// ----------------------------------- inner[b][n][tm] = states[b,n] . qs[b,tm]
__global__ void __launch_bounds__(128)
k_inner(const float* __restrict__ h, const float* __restrict__ qsf,
        float* __restrict__ inner) {
  __shared__ __attribute__((aligned(16))) _Float16 Bl[128 * LDA];   // tm-major
  __shared__ __attribute__((aligned(16))) _Float16 Al[4][16 * LDA];
  const int b  = blockIdx.y;
  const int mb = blockIdx.x >> 3;        // 0..3 (states-row blocks of 64)
  const int nb = blockIdx.x & 7;         // 0..7 (tm tiles of 128)
  const int tid = threadIdx.x;
  // B = qsflat^T slice, stored tm-major so frag loads are two aligned 16B runs.
  for (int it = 0; it < 16; ++it) {
    const int cid = tid + it * 128;
    const int j = cid >> 4, kc = (cid & 15) * 8;   // j = tm-local
    const int tm = nb * 128 + j;
    v8h hv;
    if (tm < 995) {
      const float* p = qsf + ((long)b * 995 + tm) * 128 + kc;
#pragma unroll
      for (int u = 0; u < 8; ++u) hv[u] = (_Float16)p[u];
    } else {
#pragma unroll
      for (int u = 0; u < 8; ++u) hv[u] = (_Float16)0.f;
    }
    *(v8h*)&Bl[j * LDA + kc] = hv;
  }
  const int wv = tid >> 5, lane = tid & 31;
  const int nrow0 = mb * 64 + wv * 16;
  for (int it = 0; it < 8; ++it) {
    const int cid = lane + it * 32;
    const int r = cid >> 4, c8 = (cid & 15) * 8;
    const int n = nrow0 + r;
    v8h hv;
    if (n < 199) {
      const float* p = h + ((long)b * 200 + n) * 128 + c8;
#pragma unroll
      for (int u = 0; u < 8; ++u) hv[u] = (_Float16)p[u];
    } else {
#pragma unroll
      for (int u = 0; u < 8; ++u) hv[u] = (_Float16)0.f;
    }
    *(v8h*)&Al[wv][r * LDA + c8] = hv;
  }
  __syncthreads();

  v8f zero = {0.f, 0.f, 0.f, 0.f, 0.f, 0.f, 0.f, 0.f};
  v8f acc[8];
#pragma unroll
  for (int nt = 0; nt < 8; ++nt) acc[nt] = zero;
#pragma unroll
  for (int ks = 0; ks < 4; ++ks) {
    const v16h a = frag_a_lds(&Al[wv][0], ks * 32);
    v16h bf[8];
#pragma unroll
    for (int nt = 0; nt < 8; ++nt) bf[nt] = frag_b_ldsT(Bl, ks * 32, nt * 16);
#pragma unroll
    for (int nt = 0; nt < 8; ++nt) acc[nt] = wmma_f16(a, bf[nt], acc[nt]);
  }
  const int lg = lane >> 4, nc = lane & 15;
#pragma unroll
  for (int nt = 0; nt < 8; ++nt) {
    const int tm = nb * 128 + nt * 16 + nc;
#pragma unroll
    for (int j = 0; j < 8; ++j) {
      const int n = nrow0 + 8 * lg + j;
      if (n < 199) inner[((long)b * 208 + n) * 1024 + tm] = acc[nt][j];
    }
  }
}

// ------------------------------- softmax-weighted sigmoid readout (w_b cancels)
__global__ void __launch_bounds__(128)
k_pred(const float* __restrict__ ast, const float* __restrict__ aqs,
       const float* __restrict__ inner, float* __restrict__ out) {
  const int gw = blockIdx.x * 4 + (threadIdx.x >> 5);
  if (gw >= 32 * 199) return;
  const int b = gw / 199, t = gw % 199;
  const int lane = threadIdx.x & 31;
  float aq[5];
  float aqmax = -1e30f;
#pragma unroll
  for (int m = 0; m < 5; ++m) {
    aq[m] = aqs[b * 995 + t * 5 + m];
    aqmax = fmaxf(aqmax, aq[m]);
  }
  float mloc = -1e30f;
  for (int n = lane; n <= t; n += 32) mloc = fmaxf(mloc, ast[b * 199 + n]);
#pragma unroll
  for (int o = 16; o; o >>= 1) mloc = fmaxf(mloc, __shfl_xor(mloc, o, 32));
  const float mx = mloc + aqmax;
  float se = 0.f, num = 0.f;
  for (int n = lane; n <= t; n += 32) {
    const float as = ast[b * 199 + n];
    const float* ip = &inner[((long)b * 208 + n) * 1024 + t * 5];
#pragma unroll
    for (int m = 0; m < 5; ++m) {
      const float e = __expf(as + aq[m] - mx);
      se += e;
      num += e * sigmoidf_(ip[m]);
    }
  }
#pragma unroll
  for (int o = 16; o; o >>= 1) {
    se += __shfl_xor(se, o, 32);
    num += __shfl_xor(num, o, 32);
  }
  if (lane == 0) {
    out[b * 200 + t + 1] = num / se;
    if (t == 0) out[b * 200] = 0.f;
  }
}

// ============================================================== host launcher
extern "C" void kernel_launch(void* const* d_in, const int* in_sizes, int n_in,
                              void* d_out, int out_size, void* d_ws, size_t ws_size,
                              hipStream_t stream) {
  (void)in_sizes; (void)n_in; (void)out_size; (void)ws_size;
  const int*   question  = (const int*)  d_in[0];
  const int*   response  = (const int*)  d_in[1];
  const int*   q_nb      = (const int*)  d_in[3];
  const int*   s_nb      = (const int*)  d_in[4];
  const int*   qs_skills = (const int*)  d_in[5];
  const float* emb_q     = (const float*)d_in[6];
  const float* emb_s     = (const float*)d_in[7];
  const float* emb_r     = (const float*)d_in[8];
  const float* W_ih      = (const float*)d_in[9];
  const float* b_ih      = (const float*)d_in[10];
  const float* b_hh      = (const float*)d_in[11];
  const float* ft_W      = (const float*)d_in[12];
  const float* ft_b      = (const float*)d_in[13];
  const float* agg_W     = (const float*)d_in[14];
  const float* agg_b     = (const float*)d_in[15];
  const float* last_W    = (const float*)d_in[16];
  const float* last_b    = (const float*)d_in[17];
  const float* q_W       = (const float*)d_in[18];
  const float* q_b       = (const float*)d_in[19];
  const float* k_W       = (const float*)d_in[20];
  const float* k_b       = (const float*)d_in[21];
  const float* w_W       = (const float*)d_in[22];
  float* out = (float*)d_out;

  float* ws = (float*)d_ws;
  size_t off = 0;
  auto alloc = [&](size_t n) { float* p = ws + off; off += n; return p; };
  float* Y2a  = alloc(256000UL * 128);   // hop-2 activations (131 MB)
  float* Y1a  = alloc(25600UL * 128);
  float* Y0a  = alloc(6400UL * 128);
  float* Y1b  = alloc(25600UL * 128);
  float* Y0b  = alloc(6400UL * 128);
  float* Y0c  = alloc(6400UL * 128);
  float* aggE = alloc(6400UL * 128);
  float* qtr  = alloc(6400UL * 128);
  float* Gi   = alloc(6400UL * 128);
  float* Gg   = alloc(6400UL * 128);
  float* Go   = alloc(6400UL * 128);
  float* hbuf = alloc(6400UL * 128);
  float* rcon = alloc(1024);
  float* vq   = alloc(128);
  float* vk   = alloc(128);
  float* cq   = alloc(64);
  float* ck   = alloc(64);
  float* qsf  = alloc(31840UL * 128);
  float* ast  = alloc(6400);
  float* aqs  = alloc(32000);
  float* inn  = alloc(32UL * 208 * 1024);

  const dim3 blk(128);
  auto gemm = [&](int mode, const float* X, int ldx, const float* W, int ldw,
                  const float* bias, const float* tab, int ldtab, const int* sel,
                  float* Y, int M, int act, const float* P0, const float* P1,
                  int grp) {
    k_gemm<<<dim3((M + 63) / 64), blk, 0, stream>>>(
        X, ldx, W, ldw, bias, tab, ldtab, sel, Y, M, act, mode, question, q_nb,
        s_nb, emb_q, emb_s, P0, P1, grp);
  };

  // hop aggregation: i=0
  gemm(1, nullptr, 0, agg_W + 2 * 128 * 128, 128, agg_b + 256, nullptr, 0,
       nullptr, Y2a, 256000, 1, nullptr, nullptr, 0);
  gemm(2, nullptr, 0, agg_W + 1 * 128 * 128, 128, agg_b + 128, nullptr, 0,
       nullptr, Y1a, 25600, 1, nullptr, nullptr, 0);
  gemm(3, nullptr, 0, agg_W, 128, agg_b, nullptr, 0, nullptr, Y0a, 6400, 1,
       nullptr, nullptr, 0);
  // i=1
  gemm(4, nullptr, 0, agg_W + 1 * 128 * 128, 128, agg_b + 128, nullptr, 0,
       nullptr, Y1b, 25600, 1, Y1a, Y2a, 10);
  gemm(4, nullptr, 0, agg_W, 128, agg_b, nullptr, 0, nullptr, Y0b, 6400, 1,
       Y0a, Y1a, 4);
  // i=2
  gemm(4, nullptr, 0, agg_W, 128, agg_b, nullptr, 0, nullptr, Y0c, 6400, 1,
       Y0b, Y1b, 4);
  // last + ft
  gemm(0, Y0c, 128, last_W, 128, last_b, nullptr, 0, nullptr, aggE, 6400, 1,
       nullptr, nullptr, 0);
  gemm(0, aggE, 128, ft_W, 128, ft_b, nullptr, 0, nullptr, qtr, 6400, 2,
       nullptr, nullptr, 0);

  // LSTM gates (f gate dead; emb_r side folded into rcon table)
  k_rcontrib<<<dim3(4), dim3(256), 0, stream>>>(emb_r, W_ih, b_ih, b_hh, rcon);
  gemm(0, qtr, 128, W_ih + 0,   512, nullptr, rcon + 0,   512, response, Gi, 6400, 0, nullptr, nullptr, 0);
  gemm(0, qtr, 128, W_ih + 256, 512, nullptr, rcon + 256, 512, response, Gg, 6400, 0, nullptr, nullptr, 0);
  gemm(0, qtr, 128, W_ih + 384, 512, nullptr, rcon + 384, 512, response, Go, 6400, 0, nullptr, nullptr, 0);
  k_lstm<<<dim3(3200), dim3(256), 0, stream>>>(Gi, Gg, Go, hbuf, 6400 * 128);

  // attention
  k_qsflat<<<dim3(15920), dim3(256), 0, stream>>>(question, qs_skills, emb_q,
                                                  emb_s, qsf);
  k_wvec<<<dim3(1), dim3(256), 0, stream>>>(q_W, q_b, k_W, k_b, w_W, vq, vk, cq, ck);
  k_rowdot<<<dim3((6368 + 3) / 4), blk, 0, stream>>>(hbuf, vq, cq, ast, 6368, 1);
  k_rowdot<<<dim3((31840 + 3) / 4), blk, 0, stream>>>(qsf, vk, ck, aqs, 31840, 0);
  k_inner<<<dim3(32, 32), blk, 0, stream>>>(hbuf, qsf, inn);
  k_pred<<<dim3((6368 + 3) / 4), blk, 0, stream>>>(ast, aqs, inn, out);
}
